// Encoder_87780541595717
// MI455X (gfx1250) — matrix operand order repaired and checked by reading, not compiled, not run on hardware
//
#include <hip/hip_runtime.h>

#define NL   4
#define OPT  32
#define D    3072
#define H    256
#define B    256
#define OD   (OPT * D) /* 98304 */

typedef __attribute__((ext_vector_type(16))) __bf16 v16bf;
typedef __attribute__((ext_vector_type(8)))  float  v8f;

union FragAB { v16bf v; unsigned u[8]; };
union FragC  { v8f   v; float    f[8]; };

__device__ __forceinline__ unsigned short f2bf(float f) {
    union { float f; unsigned u; } a; a.f = f;
    unsigned r = a.u + 0x7FFFu + ((a.u >> 16) & 1u);   // round-to-nearest-even
    return (unsigned short)(r >> 16);
}
__device__ __forceinline__ float bf2f(unsigned short s) {
    union { unsigned u; float f; } a; a.u = ((unsigned)s) << 16;
    return a.f;
}

// ---------------------------------------------------------------- init / prep
__global__ __launch_bounds__(256) void k_init(float* __restrict__ cur) {
    int i = blockIdx.x * 256 + threadIdx.x;
    cur[i] = 0.f;
}

__global__ __launch_bounds__(256) void k_prep(const float* __restrict__ cur,
                                              const float* __restrict__ x,
                                              float* __restrict__ tdiff,
                                              float* __restrict__ losses) {
    int i = blockIdx.x * 256 + threadIdx.x;
    tdiff[i] = cur[i] - x[i];
    if (i < B * OPT) losses[i] = 0.f;
}

// -------------------------------------------------- GEMM1: base = relu(cur@Wb+bb)
__global__ __launch_bounds__(256) void k_gemm1(const float* __restrict__ cur,
                                               const float* __restrict__ Wb,
                                               const float* __restrict__ bb,
                                               unsigned short* __restrict__ baseBf) {
    __shared__ unsigned short As[128 * 36];   // [row][k], padded stride
    __shared__ unsigned short Bs[128 * 36];   // [n][k]  (K-transposed)
    const int tid  = threadIdx.x;
    const int wid  = tid >> 5, lane = tid & 31;
    const int ln   = lane & 15, half = lane >> 4;
    const int m0   = blockIdx.y * 128, n0 = blockIdx.x * 128;

    FragC acc[8];
    for (int nt = 0; nt < 8; ++nt)
        for (int e = 0; e < 8; ++e) acc[nt].f[e] = 0.f;

    for (int kk = 0; kk < D; kk += 32) {
        for (int j = 0; j < 16; ++j) {               // A tile 128x32, fp32->bf16
            int idx = j * 256 + tid; int r = idx >> 5, c = idx & 31;
            As[r * 36 + c] = f2bf(cur[(size_t)(m0 + r) * D + kk + c]);
        }
        for (int j = 0; j < 16; ++j) {               // B tile 32x128 -> Bs[n][k]
            int idx = j * 256 + tid; int kr = idx >> 7, nc = idx & 127;
            Bs[nc * 36 + kr] = f2bf(Wb[(size_t)(kk + kr) * H + n0 + nc]);
        }
        __syncthreads();

        FragAB a;
        const int arow = wid * 16 + ln;
        for (int t = 0; t < 8; ++t) {
            int kb = ((t >> 2) << 4) + (half << 3) + ((t & 3) << 1);
            a.u[t] = *(const unsigned*)&As[arow * 36 + kb];
        }
        for (int nt = 0; nt < 8; ++nt) {
            FragAB b;
            const int brow = nt * 16 + ln;
            for (int t = 0; t < 8; ++t)
                b.u[t] = *(const unsigned*)&Bs[brow * 36 + ((t >> 2) << 4) + (half << 3) + ((t & 3) << 1)];
            acc[nt].v = __builtin_amdgcn_wmma_f32_16x16x32_bf16(
                false, a.v, false, b.v, (short)0, acc[nt].v, false, false);
        }
        __syncthreads();
    }
    for (int nt = 0; nt < 8; ++nt) {
        int gcol = n0 + nt * 16 + ln;
        float bbv = bb[gcol];
        for (int r = 0; r < 8; ++r) {
            int grow = m0 + wid * 16 + (half << 3) + r;
            float v = acc[nt].f[r] + bbv;
            v = v > 0.f ? v : 0.f;
            baseBf[grow * H + gcol] = f2bf(v);
        }
    }
}

// ------------------------------ GEMM2 + fused per-(b,option) squared-error loss
__global__ __launch_bounds__(256) void k_gemm2loss(
        const unsigned short* __restrict__ baseBf,
        const float* __restrict__ Wo,   const float* __restrict__ bo,
        const float* __restrict__ bias, const float* __restrict__ tdiff,
        float* __restrict__ losses) {
    __shared__ unsigned short As[128 * 36];
    __shared__ unsigned short Bs[128 * 36];
    const int tid  = threadIdx.x;
    const int wid  = tid >> 5, lane = tid & 31;
    const int ln   = lane & 15, half = lane >> 4;
    const int ct   = blockIdx.x;                 // 0..767
    const int o    = ct / 24, dt = ct % 24;
    const int d0   = dt * 128;
    const int col0 = o * D + d0;                 // column block inside one option
    const int m0   = blockIdx.y * 128;

    FragC acc[8];
    for (int nt = 0; nt < 8; ++nt)
        for (int e = 0; e < 8; ++e) acc[nt].f[e] = 0.f;

    for (int kk = 0; kk < H; kk += 32) {
        for (int j = 0; j < 16; ++j) {               // A already bf16
            int idx = j * 256 + tid; int r = idx >> 5, c = idx & 31;
            As[r * 36 + c] = baseBf[(m0 + r) * H + kk + c];
        }
        for (int j = 0; j < 16; ++j) {               // Wo fp32 -> bf16, transposed
            int idx = j * 256 + tid; int kr = idx >> 7, nc = idx & 127;
            Bs[nc * 36 + kr] = f2bf(Wo[(size_t)(kk + kr) * OD + col0 + nc]);
        }
        if (kk + 32 < H)                             // gfx1250 global_prefetch_b8
            __builtin_prefetch(&Wo[(size_t)(kk + 32) * OD + col0 + (tid & 127)], 0, 1);
        __syncthreads();

        FragAB a;
        const int arow = wid * 16 + ln;
        for (int t = 0; t < 8; ++t) {
            int kb = ((t >> 2) << 4) + (half << 3) + ((t & 3) << 1);
            a.u[t] = *(const unsigned*)&As[arow * 36 + kb];
        }
        for (int nt = 0; nt < 8; ++nt) {
            FragAB b;
            const int brow = nt * 16 + ln;
            for (int t = 0; t < 8; ++t)
                b.u[t] = *(const unsigned*)&Bs[brow * 36 + ((t >> 2) << 4) + (half << 3) + ((t & 3) << 1)];
            acc[nt].v = __builtin_amdgcn_wmma_f32_16x16x32_bf16(
                false, a.v, false, b.v, (short)0, acc[nt].v, false, false);
        }
        __syncthreads();
    }

    // epilogue: diff = acc + bo + bias + (cur - x); reduce squares over columns
    float coef[8];
    for (int nt = 0; nt < 8; ++nt) {
        int c = col0 + nt * 16 + ln;
        coef[nt] = bo[c] + bias[c];
    }
    for (int r = 0; r < 8; ++r) {
        int grow = m0 + wid * 16 + (half << 3) + r;
        const float* trow = tdiff + (size_t)grow * D + d0;
        float s = 0.f;
        for (int nt = 0; nt < 8; ++nt) {
            float v = acc[nt].f[r] + coef[nt] + trow[nt * 16 + ln];
            s += v * v;
        }
        s += __shfl_xor(s, 1, 32);   // reduce within each 16-lane half (wave32)
        s += __shfl_xor(s, 2, 32);
        s += __shfl_xor(s, 4, 32);
        s += __shfl_xor(s, 8, 32);
        if (ln == 0) atomicAdd(&losses[grow * OPT + o], s);
    }
}

// ---------------------------------------------------------------- argmin
__global__ __launch_bounds__(256) void k_argmin(const float* __restrict__ losses,
                                                int* __restrict__ idxbuf,
                                                float* __restrict__ encOut, int layer) {
    int b = threadIdx.x;
    const float* row = losses + b * OPT;
    float best = row[0]; int bi = 0;
    for (int o = 1; o < OPT; ++o) {
        float v = row[o];
        if (v < best) { best = v; bi = o; }
    }
    idxbuf[b] = bi;
    encOut[b * NL + layer] = (float)bi;
}

// ---------------- update: cur += base[b] . Wo[:, idx(b)*D+d] + bo + bias (gather)
__global__ __launch_bounds__(256) void k_update(
        const unsigned short* __restrict__ baseBf,
        const float* __restrict__ Wo,   const float* __restrict__ bo,
        const float* __restrict__ bias, const int* __restrict__ idxbuf,
        float* __restrict__ cur) {
    __shared__ float Ws[256 * 32];     // 256 K x 32 cols of the option's block
    __shared__ float Brow[256];
    __shared__ float Tmp[8 * 32];
    __shared__ int   rowlist[256];
    __shared__ int   cnt;
    const int tid  = threadIdx.x;
    const int o    = blockIdx.y;
    const int d0   = blockIdx.x * 32;
    const int col0 = o * D + d0;

    if (tid == 0) cnt = 0;
    __syncthreads();
    if (idxbuf[tid] == o) { int p = atomicAdd(&cnt, 1); rowlist[p] = tid; }
    __syncthreads();
    const int n = cnt;
    if (n == 0) return;                               // option unchosen: skip tile

    for (int j = 0; j < 32; ++j) {                    // stage weights once, reuse
        int idx = j * 256 + tid; int k = idx >> 5, c = idx & 31;
        Ws[k * 32 + c] = Wo[(size_t)k * OD + col0 + c];
    }
    __syncthreads();

    const int c = tid & 31, kh = tid >> 5;            // 8 K-segments x 32 cols
    for (int rr = 0; rr < n; ++rr) {
        int b = rowlist[rr];
        Brow[tid] = bf2f(baseBf[b * H + tid]);
        __syncthreads();
        float p = 0.f;
        for (int j = 0; j < 32; ++j) {
            int k = kh * 32 + j;
            p += Brow[k] * Ws[k * 32 + c];
        }
        Tmp[kh * 32 + c] = p;
        __syncthreads();
        if (tid < 32) {
            float tot = 0.f;
            for (int j = 0; j < 8; ++j) tot += Tmp[j * 32 + tid];
            int colg = col0 + tid;
            cur[(size_t)b * D + d0 + tid] += tot + bo[colg] + bias[colg];
        }
        __syncthreads();
    }
}

// ------------------------------------------------------------------- launcher
extern "C" void kernel_launch(void* const* d_in, const int* in_sizes, int n_in,
                              void* d_out, int out_size, void* d_ws, size_t ws_size,
                              hipStream_t stream) {
    const float* x    = (const float*)d_in[0];   // [B, D]
    const float* Wb   = (const float*)d_in[1];   // [D, H]
    const float* bb   = (const float*)d_in[2];   // [H]
    const float* WoS  = (const float*)d_in[3];   // [NL, H, O*D]
    const float* boS  = (const float*)d_in[4];   // [NL, O*D]
    const float* biS  = (const float*)d_in[5];   // [NL, O, D]
    float* out = (float*)d_out;                  // [B*NL enc] ++ [B*D recon]

    char* ws = (char*)d_ws;
    float* cur    = (float*)ws;                    ws += (size_t)B * D * 4;
    float* tdiff  = (float*)ws;                    ws += (size_t)B * D * 4;
    float* losses = (float*)ws;                    ws += (size_t)B * OPT * 4;
    int*   idxbuf = (int*)ws;                      ws += (size_t)B * 4;
    unsigned short* baseBf = (unsigned short*)ws;  ws += (size_t)B * H * 2;

    k_init<<<dim3((B * D) / 256), 256, 0, stream>>>(cur);
    for (int i = 0; i < NL; ++i) {
        const float* Wo   = WoS + (size_t)i * H * OD;
        const float* bo   = boS + (size_t)i * OD;
        const float* bias = biS + (size_t)i * OD;
        k_prep<<<dim3((B * D) / 256), 256, 0, stream>>>(cur, x, tdiff, losses);
        k_gemm1<<<dim3(H / 128, B / 128), 256, 0, stream>>>(cur, Wb, bb, baseBf);
        k_gemm2loss<<<dim3(OPT * (D / 128), B / 128), 256, 0, stream>>>(
            baseBf, Wo, bo, bias, tdiff, losses);
        k_argmin<<<1, 256, 0, stream>>>(losses, idxbuf, out, i);
        k_update<<<dim3(D / 32, OPT), 256, 0, stream>>>(baseBf, Wo, bo, bias, idxbuf, cur);
    }
    hipMemcpyAsync(out + B * NL, cur, (size_t)B * D * sizeof(float),
                   hipMemcpyDeviceToDevice, stream);
}
